// IAS_Net_26886495273655
// MI455X (gfx1250) — matrix operand order repaired and checked by reading, not compiled, not run on hardware
//
#include <hip/hip_runtime.h>
#include <hip/hip_bf16.h>

typedef __bf16 bf16;
typedef __attribute__((ext_vector_type(16))) __bf16 v16bf;
typedef __attribute__((ext_vector_type(8)))  __bf16 bf16x8;
typedef __attribute__((ext_vector_type(8)))  float   v8f;
typedef __attribute__((ext_vector_type(2)))  float   v2f;

#define DD   48
#define D3   110592              // 48^3
#define NC   32
#define NB   2
#define NE   7077888             // NB*NC*D3
#define PDD  52                  // padded spatial dim (halo 2 each side)
#define PD3  140608              // 52^3
#define EPSI 1e-5f

static __device__ __forceinline__ v16bf cat16(bf16x8 lo, bf16x8 hi) {
  return __builtin_shufflevector(lo, hi, 0,1,2,3,4,5,6,7,8,9,10,11,12,13,14,15);
}

// ---------------- zero fill (bf16, 8 elems/thread) ----------------
__global__ void k_zero_bf16(bf16* __restrict__ p, size_t n8) {
  size_t i = (size_t)blockIdx.x * 256 + threadIdx.x;
  if (i < n8) ((bf16x8*)p)[i] = (bf16x8)(bf16)0.0f;
}

// ---------------- channel mean over D,H,W per (b,c) ----------------
__global__ void k_channel_mean(const float* __restrict__ x, float* __restrict__ cx) {
  __shared__ float sm[256];
  int bc = blockIdx.x;
  const float* p = x + (size_t)bc * D3;
  float s = 0.f;
  for (int i = threadIdx.x; i < D3; i += 256) s += p[i];
  sm[threadIdx.x] = s; __syncthreads();
  for (int o = 128; o > 0; o >>= 1) {
    if (threadIdx.x < o) sm[threadIdx.x] += sm[threadIdx.x + o];
    __syncthreads();
  }
  if (threadIdx.x == 0) cx[bc] = sm[0] * (1.0f / (float)D3);
}

// ---------------- SE MLP ----------------
__global__ void k_mlp(const float* __restrict__ cx,
                      const float* __restrict__ w1, const float* __restrict__ b1,
                      const float* __restrict__ w2, const float* __restrict__ b2,
                      float* __restrict__ cw) {
  __shared__ float h[NB * 64];
  int t = threadIdx.x;                       // 128 threads
  int b = t >> 6, j = t & 63;
  float s = b1[j];
  for (int c = 0; c < NC; ++c) s += cx[b * NC + c] * w1[c * 64 + j];
  h[b * 64 + j] = fmaxf(s, 0.f);
  __syncthreads();
  if (t < NB * NC) {
    int bb = t >> 5, co = t & 31;
    float s2 = b2[co];
    for (int jj = 0; jj < 64; ++jj) s2 += h[bb * 64 + jj] * w2[jj * NC + co];
    cw[t] = 1.f / (1.f + __expf(-s2));
  }
}

// ------- gate + layout to padded NHWC bf16: xp[b][dz][dy][dx][ci] -------
__global__ void k_gate_pack(const float* __restrict__ x, const float* __restrict__ cw,
                            bf16* __restrict__ xp) {
  size_t t = (size_t)blockIdx.x * 256 + threadIdx.x;   // NE threads
  int c = (int)(t & 31);
  size_t r = t >> 5;                                   // b*D3 + sp
  int b  = (int)(r / D3);
  int sp = (int)(r % D3);
  int d = sp / (DD * DD), h = (sp / DD) % DD, w = sp % DD;
  float v = x[((size_t)(b * NC + c)) * D3 + sp] * cw[b * NC + c];
  size_t o = ((((size_t)b * PDD + (d + 2)) * PDD + (h + 2)) * PDD + (w + 2)) * NC + c;
  xp[o] = (bf16)v;
}

// ---- weight repack: [Cout][Cin][kd][kh][kw] f32 -> [Cout][(kd,kh,kw)][ci] bf16 ----
__global__ void k_repack_w(const float* __restrict__ w, bf16* __restrict__ out, int KVOL) {
  int i = blockIdx.x * 256 + threadIdx.x;
  int total = NC * NC * KVOL;
  if (i >= total) return;
  int co = i / (NC * KVOL);
  int r  = i % (NC * KVOL);
  int ci = r / KVOL;
  int kv = r % KVOL;
  out[co * (NC * KVOL) + kv * NC + ci] = (bf16)w[i];
}

// -------- implicit-GEMM conv via v_wmma_f32_16x16x32_bf16, branchless --------
// One wave -> 32(M) x 32(N) fp32 tile (2x2 blocking: 4 WMMAs per 8 b128 loads).
// Reduction order (kd,kh,kw,ci), ci innermost (NC==32 == one WMMA K-step).
template <int KS, int OFS>   // OFS = 2 - PAD
__global__ void k_conv_wmma(const bf16* __restrict__ xp, const bf16* __restrict__ w,
                            float* __restrict__ out) {
  const int KVOL = KS * KS * KS;
  const int KT = NC * KVOL;
  const int NT2 = D3 / 32;                   // 3456 n-tiles of width 32

  int lane = threadIdx.x & 31;
  int wave = threadIdx.x >> 5;
  int wid  = blockIdx.x * 8 + wave;          // NB*NT2 = 6912 waves
  int b  = wid / NT2;
  int nt = wid % NT2;

  int ml = lane & 15;
  int hg = lane >> 4;
  int n0 = nt * 32 + ml;
  int n1 = n0 + 16;

  const bf16* xb = xp + (size_t)b * PD3 * NC;
  const bf16* wr0 = w + (size_t)ml * KT + hg * 8;          // m-tile 0
  const bf16* wr1 = w + (size_t)(16 + ml) * KT + hg * 8;   // m-tile 1

  int od0 = n0 / (DD * DD), oh0 = (n0 / DD) % DD, ow0 = n0 % DD;
  int od1 = n1 / (DD * DD), oh1 = (n1 / DD) % DD, ow1 = n1 % DD;
  const bf16* xn0 = xb + ((size_t)((od0 + OFS) * PDD + (oh0 + OFS)) * PDD + (ow0 + OFS)) * NC + hg * 8;
  const bf16* xn1 = xb + ((size_t)((od1 + OFS) * PDD + (oh1 + OFS)) * PDD + (ow1 + OFS)) * NC + hg * 8;

  v8f acc00 = {}, acc01 = {}, acc10 = {}, acc11 = {};
  for (int kd = 0; kd < KS; ++kd) {
    for (int kh = 0; kh < KS; ++kh) {
      #pragma unroll
      for (int kw = 0; kw < KS; ++kw) {
        int kv = (kd * KS + kh) * KS + kw;
        const bf16* wp0 = wr0 + kv * NC;
        const bf16* wp1 = wr1 + kv * NC;
        v16bf a0 = cat16(*(const bf16x8*)(wp0), *(const bf16x8*)(wp0 + 16));
        v16bf a1 = cat16(*(const bf16x8*)(wp1), *(const bf16x8*)(wp1 + 16));
        size_t so = ((size_t)(kd * PDD + kh) * PDD + kw) * NC;
        const bf16* xq0 = xn0 + so;
        const bf16* xq1 = xn1 + so;
        v16bf b0 = cat16(*(const bf16x8*)(xq0), *(const bf16x8*)(xq0 + 16));
        v16bf b1 = cat16(*(const bf16x8*)(xq1), *(const bf16x8*)(xq1 + 16));
        acc00 = __builtin_amdgcn_wmma_f32_16x16x32_bf16(false, a0, false, b0, (short)0, acc00, false, false);
        acc01 = __builtin_amdgcn_wmma_f32_16x16x32_bf16(false, a0, false, b1, (short)0, acc01, false, false);
        acc10 = __builtin_amdgcn_wmma_f32_16x16x32_bf16(false, a1, false, b0, (short)0, acc10, false, false);
        acc11 = __builtin_amdgcn_wmma_f32_16x16x32_bf16(false, a1, false, b1, (short)0, acc11, false, false);
      }
    }
  }
  float* ob = out + (size_t)b * NC * D3;
  #pragma unroll
  for (int rI = 0; rI < 8; ++rI) {
    int mo0 = rI + 8 * hg;
    int mo1 = 16 + rI + 8 * hg;
    ob[(size_t)mo0 * D3 + n0] = acc00[rI];
    ob[(size_t)mo0 * D3 + n1] = acc01[rI];
    ob[(size_t)mo1 * D3 + n0] = acc10[rI];
    ob[(size_t)mo1 * D3 + n1] = acc11[rI];
  }
}

// ---------------- instance-norm stats per (b,c) ----------------
__global__ void k_in_stats(const float* __restrict__ x,
                           float* __restrict__ mean, float* __restrict__ inv) {
  __shared__ float s1[256], s2[256];
  int bc = blockIdx.x;
  const float* p = x + (size_t)bc * D3;
  float s = 0.f, q = 0.f;
  for (int i = threadIdx.x; i < D3; i += 256) { float v = p[i]; s += v; q += v * v; }
  s1[threadIdx.x] = s; s2[threadIdx.x] = q; __syncthreads();
  for (int o = 128; o > 0; o >>= 1) {
    if (threadIdx.x < o) { s1[threadIdx.x] += s1[threadIdx.x + o]; s2[threadIdx.x] += s2[threadIdx.x + o]; }
    __syncthreads();
  }
  if (threadIdx.x == 0) {
    float mu  = s1[0] * (1.0f / (float)D3);
    float var = s2[0] * (1.0f / (float)D3) - mu * mu;
    mean[bc] = mu;
    inv[bc]  = rsqrtf(var + EPSI);
  }
}

// ---- norm+relu -> Qp[bcd][m=h][k=w] row-major, K padded 48->64 (pre-zeroed) ----
__global__ void k_norm_relu_q(const float* __restrict__ x, const float* __restrict__ mean,
                              const float* __restrict__ inv, const float* __restrict__ g,
                              const float* __restrict__ be, bf16* __restrict__ Qp) {
  size_t i = (size_t)blockIdx.x * 256 + threadIdx.x;
  int bc = (int)(i / D3); int c = bc & 31;
  int sp = (int)(i % D3);
  int d = sp / (DD * DD), h = (sp / DD) % DD, w = sp % DD;
  float v = (x[i] - mean[bc]) * inv[bc] * g[c] + be[c];
  size_t bcd = (size_t)bc * DD + d;
  Qp[bcd * 3072 + h * 64 + w] = (bf16)fmaxf(v, 0.f);
}

// ---- norm+relu -> Kt[bcd][n=w][k=h] transposed, K padded (pre-zeroed) ----
__global__ void k_norm_relu_kt(const float* __restrict__ x, const float* __restrict__ mean,
                               const float* __restrict__ inv, const float* __restrict__ g,
                               const float* __restrict__ be, bf16* __restrict__ Kt) {
  size_t i = (size_t)blockIdx.x * 256 + threadIdx.x;
  int bc = (int)(i / D3); int c = bc & 31;
  int sp = (int)(i % D3);
  int d = sp / (DD * DD), h = (sp / DD) % DD, w = sp % DD;
  float v = (x[i] - mean[bc]) * inv[bc] * g[c] + be[c];
  size_t bcd = (size_t)bc * DD + d;
  Kt[bcd * 3072 + w * 64 + h] = (bf16)fmaxf(v, 0.f);
}

__global__ void k_norm_relu_f32(const float* __restrict__ x, const float* __restrict__ mean,
                                const float* __restrict__ inv, const float* __restrict__ g,
                                const float* __restrict__ be, float* __restrict__ out) {
  size_t i = (size_t)blockIdx.x * 256 + threadIdx.x;
  int bc = (int)(i / D3); int c = bc & 31;
  float v = (x[i] - mean[bc]) * inv[bc] * g[c] + be[c];
  out[i] = fmaxf(v, 0.f);
}

// -------- fuse = Q @ K per (b,c,d): one wave computes the whole 48x48 --------
// 3x3 tile blocking: per K-step, 12 b128 loads feed 9 WMMAs.
__global__ void k_fuse_wmma(const bf16* __restrict__ Qp, const bf16* __restrict__ Kt,
                            float* __restrict__ fuse) {
  int lane = threadIdx.x & 31;
  int wave = threadIdx.x >> 5;
  int bcd  = blockIdx.x * 8 + wave;          // 3072 waves total
  int ml = lane & 15, hg = lane >> 4;
  const bf16* qb = Qp + (size_t)bcd * 3072;
  const bf16* kb = Kt + (size_t)bcd * 3072;

  v8f acc[3][3] = {};
  #pragma unroll
  for (int kk = 0; kk < 64; kk += 32) {
    v16bf a[3], bm[3];
    #pragma unroll
    for (int mt = 0; mt < 3; ++mt) {
      const bf16* qr = qb + (mt * 16 + ml) * 64 + hg * 8 + kk;
      a[mt] = cat16(*(const bf16x8*)(qr), *(const bf16x8*)(qr + 16));
    }
    #pragma unroll
    for (int nt = 0; nt < 3; ++nt) {
      const bf16* kr = kb + (nt * 16 + ml) * 64 + hg * 8 + kk;
      bm[nt] = cat16(*(const bf16x8*)(kr), *(const bf16x8*)(kr + 16));
    }
    #pragma unroll
    for (int mt = 0; mt < 3; ++mt)
      #pragma unroll
      for (int nt = 0; nt < 3; ++nt)
        acc[mt][nt] = __builtin_amdgcn_wmma_f32_16x16x32_bf16(false, a[mt], false, bm[nt],
                                                              (short)0, acc[mt][nt], false, false);
  }
  float* fb = fuse + (size_t)bcd * 2304;
  #pragma unroll
  for (int mt = 0; mt < 3; ++mt)
    #pragma unroll
    for (int nt = 0; nt < 3; ++nt)
      #pragma unroll
      for (int rI = 0; rI < 8; ++rI)
        fb[(mt * 16 + rI + 8 * hg) * DD + nt * 16 + ml] = acc[mt][nt][rI];
}

// ---------------- softmax over 48^3 per (b,c), times V ----------------
__global__ void k_softmax_mul(const float* __restrict__ fuse, const float* __restrict__ V,
                              float* __restrict__ att) {
  __shared__ float red[256];
  __shared__ float sMax, sSum;
  int bc = blockIdx.x;
  const float* f = fuse + (size_t)bc * D3;
  const float* vp = V + (size_t)bc * D3;
  float* o = att + (size_t)bc * D3;

  float mx = -3.4e38f;
  for (int i = threadIdx.x; i < D3; i += 256) mx = fmaxf(mx, f[i]);
  red[threadIdx.x] = mx; __syncthreads();
  for (int t = 128; t > 0; t >>= 1) {
    if (threadIdx.x < t) red[threadIdx.x] = fmaxf(red[threadIdx.x], red[threadIdx.x + t]);
    __syncthreads();
  }
  if (threadIdx.x == 0) sMax = red[0];
  __syncthreads();
  float m0 = sMax;

  float s = 0.f;
  for (int i = threadIdx.x; i < D3; i += 256) s += __expf(f[i] - m0);
  red[threadIdx.x] = s; __syncthreads();
  for (int t = 128; t > 0; t >>= 1) {
    if (threadIdx.x < t) red[threadIdx.x] += red[threadIdx.x + t];
    __syncthreads();
  }
  if (threadIdx.x == 0) sSum = red[0];
  __syncthreads();
  float inv = 1.f / sSum;

  for (int i = threadIdx.x; i < D3; i += 256)
    o[i] = __expf(f[i] - m0) * inv * vp[i];
}

// -------- final 1x1 conv over concat(att, x): fp32 WMMA 16x16x4 -----------
// 2x1 M-blocking: shared B fragment feeds both output-channel halves.
__global__ void k_final_wmma(const float* __restrict__ att, const float* __restrict__ x,
                             const float* __restrict__ wc, const float* __restrict__ bcv,
                             float* __restrict__ out) {
  const int NT = D3 / 16;
  int lane = threadIdx.x & 31;
  int wave = threadIdx.x >> 5;
  int wid  = blockIdx.x * 8 + wave;          // NB*NT = 13824 waves
  int b  = wid / NT;
  int nt = wid % NT;
  int ml = lane & 15, hg = lane >> 4;
  int n = nt * 16 + ml;

  const float* attb = att + (size_t)b * NC * D3;
  const float* xb   = x   + (size_t)b * NC * D3;
  const float* wr0  = wc + ml * 64 + hg * 2;
  const float* wr1  = wc + (16 + ml) * 64 + hg * 2;

  v8f acc0 = {}, acc1 = {};
  #pragma unroll
  for (int kk = 0; kk < 32; kk += 4) {       // first 32 ci: att_sp
    int k = kk + hg * 2;
    v2f bfrag;
    bfrag[0] = attb[(size_t)k * D3 + n];
    bfrag[1] = attb[(size_t)(k + 1) * D3 + n];
    v2f a0 = *(const v2f*)(wr0 + kk);
    v2f a1 = *(const v2f*)(wr1 + kk);
    acc0 = __builtin_amdgcn_wmma_f32_16x16x4_f32(false, a0, false, bfrag, (short)0, acc0, false, false);
    acc1 = __builtin_amdgcn_wmma_f32_16x16x4_f32(false, a1, false, bfrag, (short)0, acc1, false, false);
  }
  #pragma unroll
  for (int kk = 32; kk < 64; kk += 4) {      // second 32 ci: original x
    int k = kk - 32 + hg * 2;
    v2f bfrag;
    bfrag[0] = xb[(size_t)k * D3 + n];
    bfrag[1] = xb[(size_t)(k + 1) * D3 + n];
    v2f a0 = *(const v2f*)(wr0 + kk);
    v2f a1 = *(const v2f*)(wr1 + kk);
    acc0 = __builtin_amdgcn_wmma_f32_16x16x4_f32(false, a0, false, bfrag, (short)0, acc0, false, false);
    acc1 = __builtin_amdgcn_wmma_f32_16x16x4_f32(false, a1, false, bfrag, (short)0, acc1, false, false);
  }
  float* ob = out + (size_t)b * NC * D3;
  #pragma unroll
  for (int rI = 0; rI < 8; ++rI) {
    int mo0 = rI + 8 * hg;
    int mo1 = 16 + rI + 8 * hg;
    ob[(size_t)mo0 * D3 + n] = acc0[rI] + bcv[mo0];
    ob[(size_t)mo1 * D3 + n] = acc1[rI] + bcv[mo1];
  }
}

extern "C" void kernel_launch(void* const* d_in, const int* in_sizes, int n_in,
                              void* d_out, int out_size, void* d_ws, size_t ws_size,
                              hipStream_t stream) {
  const float* x  = (const float*)d_in[0];
  const float* w1 = (const float*)d_in[1];
  const float* b1 = (const float*)d_in[2];
  const float* w2 = (const float*)d_in[3];
  const float* b2 = (const float*)d_in[4];
  const float* wq = (const float*)d_in[5];
  const float* gq = (const float*)d_in[6];
  const float* bq = (const float*)d_in[7];
  const float* wk = (const float*)d_in[8];
  const float* gk = (const float*)d_in[9];
  const float* bk = (const float*)d_in[10];
  const float* wv = (const float*)d_in[11];
  const float* gv = (const float*)d_in[12];
  const float* bv = (const float*)d_in[13];
  const float* wc = (const float*)d_in[14];
  const float* bc = (const float*)d_in[15];
  float* out = (float*)d_out;

  char* ws = (char*)d_ws;
  size_t off = 0;
  auto carve = [&](size_t bytes) -> char* {
    char* p = ws + off;
    off = (off + bytes + 255) & ~(size_t)255;
    return p;
  };
  float* cx    = (float*)carve(64 * 4);
  float* cw    = (float*)carve(64 * 4);
  float* meanQ = (float*)carve(64 * 4);  float* invQ = (float*)carve(64 * 4);
  float* meanK = (float*)carve(64 * 4);  float* invK = (float*)carve(64 * 4);
  float* meanV = (float*)carve(64 * 4);  float* invV = (float*)carve(64 * 4);
  const size_t XP_N = (size_t)NB * PD3 * NC;       // 8,998,912
  const size_t QP_N = (size_t)NB * NC * DD * 3072; // 9,437,184 (48x64 per mat)
  bf16*  xp    = (bf16*)carve(XP_N * 2);
  bf16*  wqb   = (bf16*)carve(27648 * 2);
  bf16*  wkb   = (bf16*)carve(128000 * 2);
  bf16*  wvb   = (bf16*)carve(1024 * 2);
  float* convQ = (float*)carve((size_t)NE * 4);    // reused as `fuse`
  float* convK = (float*)carve((size_t)NE * 4);    // reused as `att_sp`
  float* convV = (float*)carve((size_t)NE * 4);    // normalized in place -> V
  bf16*  Qp    = (bf16*)carve(QP_N * 2);
  bf16*  Kt    = (bf16*)carve(QP_N * 2);

  const int EW_BLOCKS    = NE / 256;     // 27648
  const int CONV_BLOCKS  = 864;          // 6912 waves / 8
  const int FUSE_BLOCKS  = 384;          // 3072 waves / 8
  const int FINAL_BLOCKS = 1728;         // 13824 waves / 8

  // 1) SE gating + padded NHWC pack
  k_channel_mean<<<64, 256, 0, stream>>>(x, cx);
  k_mlp<<<1, 128, 0, stream>>>(cx, w1, b1, w2, b2, cw);
  {
    size_t n8 = XP_N / 8;
    k_zero_bf16<<<(int)((n8 + 255) / 256), 256, 0, stream>>>(xp, n8);
  }
  k_gate_pack<<<EW_BLOCKS, 256, 0, stream>>>(x, cw, xp);

  // 2) weight repack -> bf16, (kd,kh,kw,ci) order
  k_repack_w<<<(27648 + 255) / 256, 256, 0, stream>>>(wq, wqb, 27);
  k_repack_w<<<(128000 + 255) / 256, 256, 0, stream>>>(wk, wkb, 125);
  k_repack_w<<<(1024 + 255) / 256, 256, 0, stream>>>(wv, wvb, 1);

  // 3) implicit-GEMM WMMA convs (OFS = 2 - PAD)
  k_conv_wmma<3, 1><<<CONV_BLOCKS, 256, 0, stream>>>(xp, wqb, convQ);
  k_conv_wmma<5, 0><<<CONV_BLOCKS, 256, 0, stream>>>(xp, wkb, convK);
  k_conv_wmma<1, 2><<<CONV_BLOCKS, 256, 0, stream>>>(xp, wvb, convV);

  // 4) instance norm + relu (Q row-major padded, K transposed padded, V f32)
  k_in_stats<<<64, 256, 0, stream>>>(convQ, meanQ, invQ);
  k_in_stats<<<64, 256, 0, stream>>>(convK, meanK, invK);
  k_in_stats<<<64, 256, 0, stream>>>(convV, meanV, invV);
  {
    size_t n8 = QP_N / 8;
    k_zero_bf16<<<(int)((n8 + 255) / 256), 256, 0, stream>>>(Qp, n8);
    k_zero_bf16<<<(int)((n8 + 255) / 256), 256, 0, stream>>>(Kt, n8);
  }
  k_norm_relu_q  <<<EW_BLOCKS, 256, 0, stream>>>(convQ, meanQ, invQ, gq, bq, Qp);
  k_norm_relu_kt <<<EW_BLOCKS, 256, 0, stream>>>(convK, meanK, invK, gk, bk, Kt);
  k_norm_relu_f32<<<EW_BLOCKS, 256, 0, stream>>>(convV, meanV, invV, gv, bv, convV);

  // 5) fuse = Q@K (convQ reused), softmax * V (convK reused)
  k_fuse_wmma<<<FUSE_BLOCKS, 256, 0, stream>>>(Qp, Kt, convQ);
  k_softmax_mul<<<64, 256, 0, stream>>>(convQ, convV, convK);

  // 6) final 1x1 conv over concat(att, x) with fp32 WMMA
  k_final_wmma<<<FINAL_BLOCKS, 256, 0, stream>>>(convK, x, wc, bc, out);
}
